// Model_27693949125306
// MI455X (gfx1250) — matrix operand order repaired and checked by reading, not compiled, not run on hardware
//
#include <hip/hip_runtime.h>
#include <hip/hip_bf16.h>
#include <math.h>

// ---------------------------------------------------------------------------
// Problem sizes (from reference): B=256, T=784, H=512, NCLS=10
// ---------------------------------------------------------------------------
#define BSZ 256
#define TSZ 784
#define HSZ 512
#define NCLS 10
#define EXPM_S 8        // scaling: A/2^8, ||skew||_2 ~ 2.3 -> scaled ~0.009
#define SCAN_BLOCKS 64  // persistent kernel: 64 blocks x 256 thr = 512 waves

// LDS row pitch (halves): 512 + 8 -> 1040B row stride = 260 dwords = 4-bank
// skew per row so 16-lane fragment reads are bank-conflict-free.
#define LDSP 520
#define BSH_HALVES (32 * LDSP)              // 32 rows of Bt per block
#define ASH_HALVES (64 * LDSP)              // 64 rows of h per block
#define SMEM_BYTES ((BSH_HALVES + ASH_HALVES) * 2)

typedef __attribute__((ext_vector_type(16))) _Float16 v16h;
typedef __attribute__((ext_vector_type(8)))  _Float16 v8h;
typedef __attribute__((ext_vector_type(8)))  float    v8f;
typedef __attribute__((ext_vector_type(2)))  float    v2f;

#if __has_builtin(__builtin_amdgcn_global_load_async_to_lds_b128) && \
    __has_builtin(__builtin_amdgcn_s_wait_asynccnt)
#define HAVE_ASYNC_LDS 1
// exact pointee type from the builtin's signature: int __vector_size__(16)
typedef int vsi4 __attribute__((vector_size(16)));
typedef __attribute__((address_space(1))) vsi4 g_vsi4;   // global int4
typedef __attribute__((address_space(3))) vsi4 l_vsi4;   // LDS int4
#else
#define HAVE_ASYNC_LDS 0
#endif

// ---------------------------------------------------------------------------
// K0: reset barrier counter, zero the f16 h ping-pong buffers, and gather the
//     permuted input xp[t][b] = inputs[b][perm[t]]   (T x B, f32)
// ---------------------------------------------------------------------------
__global__ void k_init(unsigned* __restrict__ cnt,
                       float* __restrict__ xp,
                       const float* __restrict__ inputs,
                       const int* __restrict__ perm,
                       _Float16* __restrict__ hbuf)
{
    int idx = blockIdx.x * blockDim.x + threadIdx.x;
    if (idx == 0) *cnt = 0u;
    if (idx < 2 * BSZ * HSZ) hbuf[idx] = (_Float16)0.0f;
    if (idx < TSZ * BSZ) {
        int t = idx / BSZ;
        int b = idx - t * BSZ;
        xp[idx] = inputs[b * TSZ + perm[t]];
    }
}

// ---------------------------------------------------------------------------
// K1: A0 = (triu(W,1) - triu(W,1)^T) / 2^EXPM_S
// ---------------------------------------------------------------------------
__global__ void k_skew(float* __restrict__ A0, const float* __restrict__ W)
{
    int idx = blockIdx.x * blockDim.x + threadIdx.x;
    if (idx >= HSZ * HSZ) return;
    int i = idx / HSZ, j = idx - i * HSZ;
    float v = 0.0f;
    if (j > i)      v =  W[i * HSZ + j];
    else if (j < i) v = -W[j * HSZ + i];
    A0[idx] = v * (1.0f / (float)(1 << EXPM_S));
}

// ---------------------------------------------------------------------------
// K2: E = I + A0 ;  P = A0   (start of Taylor series)
// ---------------------------------------------------------------------------
__global__ void k_initEP(float* __restrict__ E, float* __restrict__ P,
                         const float* __restrict__ A0)
{
    int idx = blockIdx.x * blockDim.x + threadIdx.x;
    if (idx >= HSZ * HSZ) return;
    int i = idx / HSZ, j = idx - i * HSZ;
    float a = A0[idx];
    P[idx] = a;
    E[idx] = a + (i == j ? 1.0f : 0.0f);
}

// ---------------------------------------------------------------------------
// K3: f32 512x512x512 GEMM on v_wmma_f32_16x16x4_f32.
//     C = alpha * (A @ Bm); if Eacc != null, Eacc += C (Taylor accumulate).
// ---------------------------------------------------------------------------
__global__ __launch_bounds__(256) void
k_gemm_f32(float* __restrict__ C, const float* __restrict__ A,
           const float* __restrict__ Bm, float alpha, float* __restrict__ Eacc)
{
    int wave = (blockIdx.x * blockDim.x + threadIdx.x) >> 5;  // 0..1023
    int lane = threadIdx.x & 31;
    int g    = lane >> 4;
    int mr   = lane & 15;
    int m0 = (wave & 31) << 4;
    int n0 = (wave >> 5) << 4;

    v8f acc = {};
    const float* arow = A + (m0 + mr) * HSZ;
    #pragma unroll 4
    for (int kc = 0; kc < HSZ / 4; ++kc) {
        int k0 = kc * 4 + 2 * g;
        v2f a = *(const v2f*)(arow + k0);
        v2f b;
        b.x = Bm[k0 * HSZ + n0 + mr];
        b.y = Bm[(k0 + 1) * HSZ + n0 + mr];
        acc = __builtin_amdgcn_wmma_f32_16x16x4_f32(
            false, a, false, b, (short)0, acc, false, false);
    }
    #pragma unroll
    for (int r = 0; r < 8; ++r) {
        int row = m0 + r + 8 * g;
        int col = n0 + mr;
        float v = alpha * acc[r];
        if (Eacc) Eacc[row * HSZ + col] += v;
        C[row * HSZ + col] = v;
    }
}

// ---------------------------------------------------------------------------
// K4: pack Borth (f32, KxN) into f16 transposed Bt[n][k]
// ---------------------------------------------------------------------------
__global__ void k_pack_bt(_Float16* __restrict__ Bt, const float* __restrict__ E)
{
    int idx = blockIdx.x * blockDim.x + threadIdx.x;
    if (idx >= HSZ * HSZ) return;
    int n = idx / HSZ, k = idx - n * HSZ;
    Bt[n * HSZ + k] = (_Float16)E[k * HSZ + n];
}

// ---------------------------------------------------------------------------
// Stage one 16B chunk global -> LDS (async on CDNA5, sync fallback).
// ---------------------------------------------------------------------------
__device__ __forceinline__ void stage16(const _Float16* gsrc, _Float16* ldst)
{
#if HAVE_ASYNC_LDS
    __builtin_amdgcn_global_load_async_to_lds_b128(
        (g_vsi4*)(void*)const_cast<_Float16*>(gsrc),  // generic -> AS1 int4*
        (l_vsi4*)(void*)ldst,                         // generic -> AS3 int4*
        0, 0);
#else
    *(uint4*)ldst = *(const uint4*)gsrc;
#endif
}

__device__ __forceinline__ void stage_wait()
{
#if HAVE_ASYNC_LDS
    __builtin_amdgcn_s_wait_asynccnt(0);
#endif
}

// ---------------------------------------------------------------------------
// K5: persistent scan. Grid remap: block = (m_quad: 4 M-tiles, n_pair: 2
//     N-tiles); 8 waves each own a 16x16 tile. Bt rows for the block's two
//     N-tiles (32 x 512 f16) are staged into LDS ONCE; the dynamic h operand
//     (64 x 512 f16) is staged per step via async global->LDS. All fragment
//     reads are ds_load_b128; math is v_wmma_f32_16x16x32_f16; one grid-wide
//     monotonic atomic barrier per step; h ping-pongs in global.
// ---------------------------------------------------------------------------
__global__ __launch_bounds__(256, 1) void
k_scan(const _Float16* __restrict__ Bt,
       _Float16* __restrict__ hbuf,
       const float* __restrict__ xp,
       const float* __restrict__ W_in,
       const float* __restrict__ b_mod,
       unsigned* __restrict__ cnt)
{
    extern __shared__ _Float16 smem[];
    _Float16* Bsh = smem;                 // 32 rows x LDSP
    _Float16* Ash = smem + BSH_HALVES;    // 64 rows x LDSP

    const int tid    = threadIdx.x;
    const int w      = tid >> 5;          // wave in block: 0..7
    const int lane   = tid & 31;
    const int g      = lane >> 4;
    const int mr     = lane & 15;
    const int m_quad = blockIdx.x & 3;    // 4 quads x 4 M-tiles = 16 M-tiles
    const int n_pair = blockIdx.x >> 2;   // 16 pairs x 2 N-tiles = 32 N-tiles
    const int m_sub  = w & 3;             // M-tile within quad
    const int n_sub  = w >> 2;            // N-tile within pair

    const int m_blk0 = m_quad * 64;                   // block's first h row
    const int m_loc0 = m_sub * 16;                    // wave's first local row
    const int n      = (n_pair * 2 + n_sub) * 16 + mr;  // lane's column

    const float win = W_in[n];
    const float bm  = b_mod[n];

    // ---- stage Bt rows [n_pair*32, n_pair*32+32) into LDS, once ----
    {
        const _Float16* bsrc = Bt + (size_t)(n_pair * 32) * HSZ;
        #pragma unroll
        for (int i = 0; i < 8; ++i) {                 // 2048 chunks / 256 thr
            int c   = tid + i * 256;
            int row = c >> 6, col = (c & 63) * 8;
            *(uint4*)(Bsh + row * LDSP + col) =
                *(const uint4*)(bsrc + row * HSZ + col);
        }
        __syncthreads();
    }

    const _Float16* bls = Bsh + (n_sub * 16 + mr) * LDSP;   // lane's B row
    const _Float16* als = Ash + (m_loc0 + mr) * LDSP;       // lane's A row

    for (int t = 0; t < TSZ; ++t) {
        const _Float16* hin  = hbuf + (t & 1) * (BSZ * HSZ);
        _Float16*       hout = hbuf + ((t + 1) & 1) * (BSZ * HSZ);

        // ---- stage h rows [m_blk0, m_blk0+64) into LDS (async DMA) ----
        const _Float16* asrc = hin + (size_t)m_blk0 * HSZ;
        #pragma unroll
        for (int i = 0; i < 16; ++i) {                // 4096 chunks / 256 thr
            int c   = tid + i * 256;
            int row = c >> 6, col = (c & 63) * 8;
            stage16(asrc + row * HSZ + col, Ash + row * LDSP + col);
        }
        stage_wait();
        __syncthreads();

        // ---- 16x16 tile GEMM over K=512 with f16 WMMA from LDS ----
        v8f acc = {};
        #pragma unroll 4
        for (int kc = 0; kc < HSZ / 32; ++kc) {
            int kb = kc * 32;
            v8h alo = *(const v8h*)(als + kb + 8 * g);
            v8h ahi = *(const v8h*)(als + kb + 16 + 8 * g);
            v16h a;
            #pragma unroll
            for (int i = 0; i < 8; ++i) { a[i] = alo[i]; a[i + 8] = ahi[i]; }
            v8h blo = *(const v8h*)(bls + kb + 8 * g);
            v8h bhi = *(const v8h*)(bls + kb + 16 + 8 * g);
            v16h b;
            #pragma unroll
            for (int i = 0; i < 8; ++i) { b[i] = blo[i]; b[i + 8] = bhi[i]; }
            acc = __builtin_amdgcn_wmma_f32_16x16x32_f16(
                false, a, false, b, (short)0, acc, false, false);
        }

        // ---- epilogue: pre = acc + xp[t][m]*W_in[n]; signed-shrink act ----
        #pragma unroll
        for (int r = 0; r < 8; ++r) {
            int m = m_blk0 + m_loc0 + r + 8 * g;
            float pre = acc[r] + xp[t * BSZ + m] * win;
            float ab  = fabsf(pre) + bm;
            float hv  = (ab > 0.0f) ? ((pre < 0.0f) ? -ab : ab) : 0.0f;
            hout[m * HSZ + n] = (_Float16)hv;
        }

        // ---- grid-wide barrier (monotonic counter) ----
        __threadfence();          // every lane: h stores device-visible
        __syncthreads();
        if (tid == 0) {
            __hip_atomic_fetch_add(cnt, 1u, __ATOMIC_RELEASE,
                                   __HIP_MEMORY_SCOPE_AGENT);
            unsigned target = (unsigned)(t + 1) * (unsigned)gridDim.x;
            while (__hip_atomic_load(cnt, __ATOMIC_ACQUIRE,
                                     __HIP_MEMORY_SCOPE_AGENT) < target)
                __builtin_amdgcn_s_sleep(2);
        }
        __syncthreads();
    }
}

// ---------------------------------------------------------------------------
// K6: out[b][c] = sum_h hT[b][h] * W_lin[c][h] + b_lin[c]   (256x10, tiny)
// ---------------------------------------------------------------------------
__global__ void k_out(float* __restrict__ out,
                      const _Float16* __restrict__ hT,
                      const float* __restrict__ W_lin,
                      const float* __restrict__ b_lin)
{
    int idx = blockIdx.x * blockDim.x + threadIdx.x;
    if (idx >= BSZ * NCLS) return;
    int b = idx / NCLS, c = idx - b * NCLS;
    const _Float16* hrow = hT + b * HSZ;
    const float*    wrow = W_lin + c * HSZ;
    float s = 0.0f;
    #pragma unroll 8
    for (int h = 0; h < HSZ; ++h) s += (float)hrow[h] * wrow[h];
    out[idx] = s + b_lin[c];
}

// ---------------------------------------------------------------------------
// Host side
// ---------------------------------------------------------------------------
extern "C" void kernel_launch(void* const* d_in, const int* in_sizes, int n_in,
                              void* d_out, int out_size, void* d_ws, size_t ws_size,
                              hipStream_t stream)
{
    const float* inputs = (const float*)d_in[0];   // (256, 784)
    const int*   perm   = (const int*)d_in[1];     // (784,) int32
    const float* W_skew = (const float*)d_in[2];   // (512, 512)
    const float* W_in   = (const float*)d_in[3];   // (512, 1)
    const float* b_mod  = (const float*)d_in[4];   // (512,)
    const float* W_lin  = (const float*)d_in[5];   // (10, 512)
    const float* b_lin  = (const float*)d_in[6];   // (10,)
    float* out = (float*)d_out;

    char* ws = (char*)d_ws;
    size_t off = 0;
    auto carve = [&](size_t bytes) -> void* {
        off = (off + 255) & ~(size_t)255;
        void* p = ws + off;
        off += bytes;
        return p;
    };
    unsigned*  cnt  = (unsigned*)carve(256);
    float*     xp   = (float*)carve((size_t)TSZ * BSZ * 4);
    float*     A0   = (float*)carve((size_t)HSZ * HSZ * 4);
    float*     P0   = (float*)carve((size_t)HSZ * HSZ * 4);
    float*     P1   = (float*)carve((size_t)HSZ * HSZ * 4);
    float*     E0   = (float*)carve((size_t)HSZ * HSZ * 4);
    float*     E1   = (float*)carve((size_t)HSZ * HSZ * 4);
    _Float16*  Bt   = (_Float16*)carve((size_t)HSZ * HSZ * 2);
    _Float16*  hbuf = (_Float16*)carve((size_t)2 * BSZ * HSZ * 2);

    k_init<<<(2 * BSZ * HSZ + 255) / 256, 256, 0, stream>>>(cnt, xp, inputs, perm, hbuf);
    k_skew<<<(HSZ * HSZ + 255) / 256, 256, 0, stream>>>(A0, W_skew);
    k_initEP<<<(HSZ * HSZ + 255) / 256, 256, 0, stream>>>(E0, P0, A0);

    const int GEMM_BLOCKS = (HSZ / 16) * (HSZ / 16) / 8;  // 128
    float* Pin = P0; float* Pout = P1;
    for (int k = 2; k <= 10; ++k) {   // Taylor: P = P@A0/k ; E += P
        k_gemm_f32<<<GEMM_BLOCKS, 256, 0, stream>>>(Pout, Pin, A0, 1.0f / (float)k, E0);
        float* tmp = Pin; Pin = Pout; Pout = tmp;
    }
    float* Ein = E0; float* Eout = E1;
    for (int s = 0; s < EXPM_S; ++s) {  // squaring: E = E@E
        k_gemm_f32<<<GEMM_BLOCKS, 256, 0, stream>>>(Eout, Ein, Ein, 1.0f, nullptr);
        float* tmp = Ein; Ein = Eout; Eout = tmp;
    }

    k_pack_bt<<<(HSZ * HSZ + 255) / 256, 256, 0, stream>>>(Bt, Ein);

    k_scan<<<SCAN_BLOCKS, 256, SMEM_BYTES, stream>>>(Bt, hbuf, xp, W_in, b_mod, cnt);

    k_out<<<(BSZ * NCLS + 255) / 256, 256, 0, stream>>>(out, hbuf, W_lin, b_lin);
}
